// CodebookEMA_83734682403544
// MI455X (gfx1250) — matrix operand order repaired and checked by reading, not compiled, not run on hardware
//
#include <hip/hip_runtime.h>
#include <hip/hip_bf16.h>

typedef __attribute__((ext_vector_type(16))) __bf16 v16bf;
typedef __attribute__((ext_vector_type(8)))  float  v8f;

#define K_CODES 1024
#define DDIM    64
#define NVEC    131072          // 32*64*64 vectors
#define NELEM   8388608         // NVEC * DDIM
#define BETA    0.25f
#define DECAY   0.99f
#define EPSV    1e-5f

// ---- workspace layout (float offsets) ----
#define OFF_EMBED_SUM 0         // [K*D]   fp32 accum
#define OFF_COUNTS    65536     // [K]     fp32 accum
#define OFF_LOSS      66560     // [1]     fp32 accum
#define OFF_W2        66564     // [K]     ||w||^2
#define OFF_IDX       67588     // [N]     int32
#define OFF_BBUF      198664    // [64K bf16] B-operand swizzled (32B aligned)
#define ZERO_COUNT    66561     // embed_sum + counts + loss

// ---- d_out layout (float offsets, tuple return order) ----
#define OUT_ZQ   0              // [32,64,64,64]
#define OUT_IDX  8388608        // [N] (as float)
#define OUT_LOSS 8519680        // scalar
#define OUT_EMB  8519681        // [K,D]
#define OUT_CS   8585217        // [K]
#define OUT_EA   8586241        // [K,D]

// 16B vector type + address-space-qualified pointers for the async-DMA builtin
typedef int async_v4i __attribute__((vector_size(16)));
typedef __attribute__((address_space(1))) async_v4i* async_gptr;   // global
typedef __attribute__((address_space(3))) async_v4i* async_lptr;   // LDS

__device__ __forceinline__ unsigned short f2bf_bits(float f) {
    union { float f; unsigned u; } v; v.f = f;
    unsigned r = v.u + 0x7FFFu + ((v.u >> 16) & 1u);   // round-to-nearest-even
    return (unsigned short)(r >> 16);
}
__device__ __forceinline__ __bf16 f2bf(float f) {
    union { unsigned short s; __bf16 b; } o; o.s = f2bf_bits(f);
    return o.b;
}

// async 16B global -> LDS copy (ASYNCcnt-tracked DMA, no VGPR round trip)
__device__ __forceinline__ void async_cp16(const unsigned char* g, unsigned char* l) {
#if __has_builtin(__builtin_amdgcn_global_load_async_to_lds_b128)
    __builtin_amdgcn_global_load_async_to_lds_b128(
        (async_gptr)(void*)g, (async_lptr)(void*)l, 0, 0);
#else
    unsigned lo = (unsigned)(size_t)(__attribute__((address_space(3))) unsigned char*)l;
    unsigned long long ga = (unsigned long long)(size_t)g;
    asm volatile("global_load_async_to_lds_b128 %0, %1, off"
                 :: "v"(lo), "v"(ga) : "memory");
#endif
}
__device__ __forceinline__ void wait_async0() {
#if __has_builtin(__builtin_amdgcn_s_wait_asynccnt)
    __builtin_amdgcn_s_wait_asynccnt(0);
#else
    asm volatile("s_wait_asynccnt 0x0" ::: "memory");
#endif
}

// ---------------- zero accumulators ----------------
__global__ void k_zero(float* ws) {
    int t = blockIdx.x * blockDim.x + threadIdx.x;
    if (t < ZERO_COUNT) ws[t] = 0.0f;
}

// ------- prep: bf16 B-operand swizzle + ||w||^2 -------
// Bbuf tile layout: [g=0..63 col tiles][p=0..1 k-halves][lane=0..31] x 16 bf16.
// B element e of lane L maps to k = e + 16*(L/16), code = g*16 + (L%16).
__global__ void __launch_bounds__(256) k_prep(const float* __restrict__ emb,
                                              float* __restrict__ ws) {
    int t = blockIdx.x * 256 + threadIdx.x;       // 0..4095
    int lane = t & 31;
    int gp   = t >> 5;                            // g*2 + p
    int p    = gp & 1;
    int g    = gp >> 1;
    int code = g * 16 + (lane & 15);
    int hh   = lane >> 4;
    int dbase = p * 32 + 16 * hh;
    unsigned short* bbuf = (unsigned short*)(ws + OFF_BBUF);
    const float* src = emb + code * DDIM + dbase;  // 16 contiguous floats
    #pragma unroll
    for (int e = 0; e < 16; ++e) bbuf[t * 16 + e] = f2bf_bits(src[e]);

    if (t < K_CODES) {
        const float* w = emb + t * DDIM;
        float s = 0.0f;
        #pragma unroll 8
        for (int d = 0; d < DDIM; ++d) s += w[d] * w[d];
        ws[OFF_W2 + t] = s;
    }
}

// ------- assign: fused distance-GEMM (bf16 WMMA) + argmin -------
// One wave per 16 rows. score[n,k] = w2[k] - 2*x.w  (x^2 dropped: row-constant).
// Codebook staged through LDS in two 64KB chunks via async global->LDS DMA,
// shared by all 8 waves of the block (8x cut in L2 read traffic).
__global__ void __launch_bounds__(256) k_assign(const float* __restrict__ z,
                                                const float* __restrict__ ws,
                                                int* __restrict__ idx_ws) {
    __shared__ __align__(128) unsigned char smemB[65536];   // 32 col-tiles of B

    const int lane = threadIdx.x & 31;
    const int wave = threadIdx.x >> 5;
    const int tile = blockIdx.x * 8 + wave;       // 0..8191
    const int nbase = tile * 16;
    const int row = lane & 15;
    const int hh  = lane >> 4;
    const int n   = nbase + row;
    const int b   = n >> 12;                      // n = b*4096 + h*64 + w
    const int rem = n & 4095;
    const int zbase = b * 262144 + rem;           // + c*4096 walks channels

    // A operand (16x64, scaled by -2), contiguous 16-bit A layout:
    // element e -> k = e + 16*h
    v16bf a0, a1;
    #pragma unroll
    for (int e = 0; e < 16; ++e) {
        int k0 = e + 16 * hh;
        a0[e] = f2bf(-2.0f * z[zbase + k0 * 4096]);
        a1[e] = f2bf(-2.0f * z[zbase + (k0 + 32) * 4096]);
    }

    const unsigned char* gB = (const unsigned char*)(ws + OFF_BBUF);
    const float* w2 = ws + OFF_W2;
    const int col = lane & 15;

    float bestv[8]; int bestc[8];
    #pragma unroll
    for (int j = 0; j < 8; ++j) { bestv[j] = 3.4e38f; bestc[j] = 0; }

    for (int ch = 0; ch < 2; ++ch) {
        __syncthreads();   // prev chunk fully consumed before overwrite
        // stage 64KB: 256 threads x 16 iters x 16B async DMA
        #pragma unroll
        for (int i = 0; i < 16; ++i) {
            int off = (i * 256 + (int)threadIdx.x) * 16;
            async_cp16(gB + ch * 65536 + off, smemB + off);
        }
        wait_async0();     // my wave's DMAs landed in LDS
        __syncthreads();   // everyone's DMAs landed

        const v16bf* bl = (const v16bf*)smemB;
        for (int gl = 0; gl < 32; ++gl) {
            v16bf b0 = bl[(gl * 2 + 0) * 32 + lane];
            v16bf b1 = bl[(gl * 2 + 1) * 32 + lane];
            const int g = ch * 32 + gl;
            float wsq = w2[g * 16 + col];
            v8f c = {wsq, wsq, wsq, wsq, wsq, wsq, wsq, wsq};
            c = __builtin_amdgcn_wmma_f32_16x16x32_bf16(false, a0, false, b0,
                                                        (short)0, c, false, false);
            c = __builtin_amdgcn_wmma_f32_16x16x32_bf16(false, a1, false, b1,
                                                        (short)0, c, false, false);
            const int cc = g * 16 + col;
            #pragma unroll
            for (int j = 0; j < 8; ++j)
                if (c[j] < bestv[j]) { bestv[j] = c[j]; bestc[j] = cc; }
        }
    }

    // reduce across the 16 lanes sharing each row (stays within wave32 halves)
    #pragma unroll
    for (int off = 1; off < 16; off <<= 1) {
        #pragma unroll
        for (int j = 0; j < 8; ++j) {
            float ov = __shfl_xor(bestv[j], off, 32);
            int   oc = __shfl_xor(bestc[j], off, 32);
            if (ov < bestv[j] || (ov == bestv[j] && oc < bestc[j])) {
                bestv[j] = ov; bestc[j] = oc;
            }
        }
    }
    if (col == 0) {  // lanes 0 (rows 0..7) and 16 (rows 8..15)
        #pragma unroll
        for (int j = 0; j < 8; ++j)
            idx_ws[nbase + j + 8 * hh] = bestc[j];
    }
}

// ------- scatter: z_q output, loss, EMA statistics -------
__global__ void __launch_bounds__(256) k_scatter(const float* __restrict__ z,
                                                 const float* __restrict__ emb,
                                                 const int* __restrict__ idx_ws,
                                                 float* __restrict__ ws,
                                                 float* __restrict__ out) {
    const int t = blockIdx.x * 256 + threadIdx.x;  // NCHW flat index
    const int n = ((t >> 18) << 12) | (t & 4095);  // b*4096 + h*64 + w
    const int c = (t >> 12) & 63;
    const int code = idx_ws[n];
    const float zq = emb[code * DDIM + c];
    const float zv = z[t];
    out[OUT_ZQ + t] = zq;                          // STE value == z_q
    atomicAdd(ws + OFF_EMBED_SUM + code * DDIM + c, zv);
    if (c == 0) {
        atomicAdd(ws + OFF_COUNTS + code, 1.0f);
        out[OUT_IDX + n] = (float)code;
    }
    const float d = zv - zq;
    __shared__ float red[256];
    red[threadIdx.x] = d * d;
    __syncthreads();
    for (int s = 128; s > 0; s >>= 1) {
        if (threadIdx.x < s) red[threadIdx.x] += red[threadIdx.x + s];
        __syncthreads();
    }
    if (threadIdx.x == 0) atomicAdd(ws + OFF_LOSS, red[0]);
}

// ------- finalize: EMA update + smoothing + loss scale -------
__global__ void __launch_bounds__(1024) k_finalize(const float* __restrict__ cluster_size,
                                                   const float* __restrict__ embed_avg,
                                                   const float* __restrict__ ws,
                                                   float* __restrict__ out) {
    __shared__ float red[1024];
    const int k = threadIdx.x;
    const float ncs = cluster_size[k] * DECAY + (1.0f - DECAY) * ws[OFF_COUNTS + k];
    out[OUT_CS + k] = ncs;
    red[k] = ncs;
    __syncthreads();
    for (int s = 512; s > 0; s >>= 1) {
        if (k < s) red[k] += red[k + s];
        __syncthreads();
    }
    const float ntot = red[0];
    const float smoothed = (ncs + EPSV) / (ntot + (float)K_CODES * EPSV) * ntot;
    const float inv = 1.0f / smoothed;
    for (int d = 0; d < DDIM; ++d) {
        const float ea = embed_avg[k * DDIM + d] * DECAY +
                         (1.0f - DECAY) * ws[OFF_EMBED_SUM + k * DDIM + d];
        out[OUT_EA  + k * DDIM + d] = ea;
        out[OUT_EMB + k * DDIM + d] = ea * inv;
    }
    if (k == 0) out[OUT_LOSS] = ws[OFF_LOSS] * (BETA / (float)NELEM);
}

extern "C" void kernel_launch(void* const* d_in, const int* in_sizes, int n_in,
                              void* d_out, int out_size, void* d_ws, size_t ws_size,
                              hipStream_t stream) {
    const float* z   = (const float*)d_in[0];
    const float* emb = (const float*)d_in[1];
    const float* cs  = (const float*)d_in[2];
    const float* ea  = (const float*)d_in[3];
    float* ws  = (float*)d_ws;
    float* out = (float*)d_out;
    int* idx_ws = (int*)(ws + OFF_IDX);

    k_zero    <<<(ZERO_COUNT + 255) / 256, 256, 0, stream>>>(ws);
    k_prep    <<<16,    256,  0, stream>>>(emb, ws);
    k_assign  <<<1024,  256,  0, stream>>>(z, ws, idx_ws);
    k_scatter <<<NELEM / 256, 256, 0, stream>>>(z, emb, idx_ws, ws, out);
    k_finalize<<<1,     1024, 0, stream>>>(cs, ea, ws, out);
}
